// get_model_6562710028516
// MI455X (gfx1250) — matrix operand order repaired and checked by reading, not compiled, not run on hardware
//
#include <hip/hip_runtime.h>

typedef _Float16 v16h __attribute__((ext_vector_type(16)));
typedef float    v8f  __attribute__((ext_vector_type(8)));

struct LayerDesc {
    const _Float16* W;   // [CoutPad][Kld] f16, zero-padded
    const float*    s;   // [CoutPad] folded bn scale (0 in pad)
    const float*    t;   // [CoutPad] folded bn shift (0 in pad)
    int Kld;             // round_up(Cin, 32)
    int CoutPad;         // round_up(Cout, 32) == Kld of next layer
    int CoutReal;
};

static constexpr int BATCH = 8;
static constexpr int NPTS  = 4096;

// ---------------- WMMA fragment loaders (contiguous K order k = 16*laneHi + s) ------------

union FragU { v16h v; unsigned u[8]; };

// A-matrix 16x32 f16: lane holds row M=lane%16; 16 contiguous K per lane half.
// p must already point at (row*ld + kb + laneHi*16)
__device__ inline v16h load_a16(const _Float16* p) {
    FragU f;
    const unsigned* q0 = (const unsigned*)(p);
    const unsigned* q1 = (const unsigned*)(p + 8);
    f.u[0] = q0[0]; f.u[1] = q0[1]; f.u[2] = q0[2]; f.u[3] = q0[3];
    f.u[4] = q1[0]; f.u[5] = q1[1]; f.u[6] = q1[2]; f.u[7] = q1[3];
    return f.v;
}

// B-matrix 32x16 f16: lane holds col N=lane%16; 16 contiguous K per lane half.
// p must already point at (col*Kld + kb + laneHi*16)
__device__ inline v16h load_b16(const _Float16* p) {
    FragU f;
    const unsigned* q = (const unsigned*)p;
#pragma unroll
    for (int i = 0; i < 8; ++i) f.u[i] = q[i];
    return f.v;
}

__device__ inline v8f wmma16(v16h a, v16h b, v8f c) {
    return __builtin_amdgcn_wmma_f32_16x16x32_f16(false, a, false, b,
                                                  (short)0, c, false, false);
}

// One fused conv layer: out[k x CoutPad] = relu(in[k x Kld] @ W^T * s + t), f16 out.
// Register-blocked 1Mx2N per wave: each A fragment feeds two WMMAs.
__device__ inline void mlp_layer(const _Float16* in, int ldi, _Float16* out, int ldo,
                                 const LayerDesc L, int k) {
    int wave = threadIdx.x >> 5, lane = threadIdx.x & 31;
    int laneHi = lane >> 4, lane16 = lane & 15;
    int nwaves = blockDim.x >> 5;
    int mt = k >> 4, ntp = L.CoutPad >> 5;   // pairs of N tiles (CoutPad % 32 == 0)
    for (int pIdx = wave; pIdx < mt * ntp; pIdx += nwaves) {
        int tm = pIdx / ntp, tn0 = (pIdx - tm * ntp) * 2;
        v8f acc0 = {}, acc1 = {};
        const _Float16* ap  = in + (size_t)(tm * 16 + lane16) * ldi + laneHi * 16;
        const _Float16* bp0 = L.W + (size_t)(tn0 * 16 + lane16) * L.Kld + laneHi * 16;
        const _Float16* bp1 = bp0 + (size_t)16 * L.Kld;
        for (int kb = 0; kb < L.Kld; kb += 32) {
            __builtin_prefetch(bp0 + kb + 32, 0, 1);
            __builtin_prefetch(bp1 + kb + 32, 0, 1);
            v16h av  = load_a16(ap + kb);
            v16h bv0 = load_b16(bp0 + kb);
            v16h bv1 = load_b16(bp1 + kb);
            acc0 = wmma16(av, bv0, acc0);
            acc1 = wmma16(av, bv1, acc1);
        }
        int col0 = tn0 * 16 + lane16, col1 = col0 + 16;
        float s0 = L.s[col0], t0 = L.t[col0];
        float s1 = L.s[col1], t1 = L.t[col1];
#pragma unroll
        for (int r = 0; r < 8; ++r) {
            int row = tm * 16 + r + laneHi * 8;
            out[(size_t)row * ldo + col0] = (_Float16)fmaxf(acc0[r] * s0 + t0, 0.f);
            out[(size_t)row * ldo + col1] = (_Float16)fmaxf(acc1[r] * s1 + t1, 0.f);
        }
    }
}

// ------------- Fused: ball-query -> gather -> 3-layer MLP -> maxpool --------
// One block per (b, s) group; 128 threads (4 waves). Activations live in LDS.
__global__ void k_group_mlp(const float* __restrict__ xyz,      // (B, n, 3)
                            const float* __restrict__ newxyz,   // (B, npoint, 3)
                            const float* __restrict__ feats,    // (B, n, pc)
                            int n, int npoint, int k, int pc, float r2,
                            LayerDesc L0, LayerDesc L1, LayerDesc L2,
                            float* __restrict__ outFeat, int outStride, int chOff,
                            int sA, int sB) {
    extern __shared__ char smem[];
    _Float16* bufA = (_Float16*)smem;
    _Float16* bufB = bufA + (size_t)k * sA;
    int* idxArr = (int*)(bufB + (size_t)k * sB);

    int tid = threadIdx.x;
    int bs = blockIdx.x;
    int b = bs / npoint;
    float cx = newxyz[bs * 3 + 0], cy = newxyz[bs * 3 + 1], cz = newxyz[bs * 3 + 2];
    const float* xb = xyz + (size_t)b * n * 3;

    // ---- ball query: first-k ascending indices within radius (wave 0 only)
    if (tid < 32) {
        int cnt = 0, first = 0;
        for (int base = 0; base < n; base += 32) {
            int j = base + tid;
            float dx = xb[j * 3] - cx, dy = xb[j * 3 + 1] - cy, dz = xb[j * 3 + 2] - cz;
            bool p = (dx * dx + dy * dy + dz * dz) <= r2;
            unsigned m = (unsigned)__ballot(p);
            if (tid == 0) {
                while (m && cnt < k) {
                    int bit = __builtin_ctz(m);
                    m &= m - 1;
                    if (cnt == 0) first = base + bit;
                    idxArr[cnt++] = base + bit;
                }
            }
            cnt = __shfl(cnt, 0);
            if (cnt >= k) break;
        }
        first = __shfl(first, 0);
        if (cnt == 0) { cnt = 1; if (tid == 0) idxArr[0] = first; }
        for (int i = cnt + tid; i < k; i += 32) idxArr[i] = first;
    }
    __syncthreads();

    // ---- gather grouped features [pc feats | 3 rel-xyz | zero pad] as f16
    int Kld0 = L0.Kld;
    for (int u = tid; u < k * Kld0; u += blockDim.x) {
        int row = u / Kld0, c = u - row * Kld0;
        int j = idxArr[row];
        float v = 0.f;
        if (c < pc) v = feats[((size_t)b * n + j) * pc + c];
        else if (c < pc + 3) {
            int d = c - pc;
            v = xb[j * 3 + d] - (d == 0 ? cx : (d == 1 ? cy : cz));
        }
        bufA[(size_t)row * sA + c] = (_Float16)v;
    }
    __syncthreads();

    // ---- 3 conv layers, ping-pong A->B->A->B
    // CoutPad is a multiple of 32 and equals the next layer's Kld; pad channels
    // produce exact zeros (zero weights, s=t=0), so no inter-layer zero-fill.
    mlp_layer(bufA, sA, bufB, sB, L0, k);
    __syncthreads();
    mlp_layer(bufB, sB, bufA, sA, L1, k);
    __syncthreads();
    mlp_layer(bufA, sA, bufB, sB, L2, k);
    __syncthreads();

    // ---- maxpool over k neighbors
    for (int c = tid; c < L2.CoutReal; c += blockDim.x) {
        float m = -3.4e38f;
        for (int r = 0; r < k; ++r) m = fmaxf(m, (float)bufB[(size_t)r * sB + c]);
        outFeat[(size_t)bs * outStride + chOff + c] = m;
    }
}

// ---------------- Generic WMMA GEMM (A in global f16) -----------------------
// Y = relu((A @ W^T) * s + t); optional f16 and/or f32 outputs.
// One wave per 16M x 32N tile pair (A fragment reused for two WMMAs).
__global__ void k_gemm(const _Float16* __restrict__ A, int lda, int M,
                       LayerDesc L,
                       _Float16* __restrict__ out16, int ldo,
                       float* __restrict__ out32, int stride32, int chOff) {
    int wave = threadIdx.x >> 5, lane = threadIdx.x & 31;
    int laneHi = lane >> 4, lane16 = lane & 15;
    int mt = M >> 4, ntp = L.CoutPad >> 5;
    int pIdx = blockIdx.x * 4 + wave;
    if (pIdx >= mt * ntp) return;
    int tm = pIdx / ntp, tn0 = (pIdx - tm * ntp) * 2;
    v8f acc0 = {}, acc1 = {};
    const _Float16* ap  = A + (size_t)(tm * 16 + lane16) * lda + laneHi * 16;
    const _Float16* bp0 = L.W + (size_t)(tn0 * 16 + lane16) * L.Kld + laneHi * 16;
    const _Float16* bp1 = bp0 + (size_t)16 * L.Kld;
    for (int kb = 0; kb < L.Kld; kb += 32) {
        __builtin_prefetch(bp0 + kb + 32, 0, 1);
        __builtin_prefetch(bp1 + kb + 32, 0, 1);
        v16h av  = load_a16(ap + kb);
        v16h bv0 = load_b16(bp0 + kb);
        v16h bv1 = load_b16(bp1 + kb);
        acc0 = wmma16(av, bv0, acc0);
        acc1 = wmma16(av, bv1, acc1);
    }
    int col0 = tn0 * 16 + lane16, col1 = col0 + 16;
    float s0 = L.s[col0], t0 = L.t[col0];
    float s1 = L.s[col1], t1 = L.t[col1];
#pragma unroll
    for (int r = 0; r < 8; ++r) {
        int row = tm * 16 + r + laneHi * 8;
        float v0 = fmaxf(acc0[r] * s0 + t0, 0.f);
        float v1 = fmaxf(acc1[r] * s1 + t1, 0.f);
        if (out16) {
            out16[(size_t)row * ldo + col0] = (_Float16)v0;
            out16[(size_t)row * ldo + col1] = (_Float16)v1;
        }
        if (out32) {
            if (col0 < L.CoutReal) out32[(size_t)row * stride32 + chOff + col0] = v0;
            if (col1 < L.CoutReal) out32[(size_t)row * stride32 + chOff + col1] = v1;
        }
    }
}

// ---------------- Scalar / support kernels ----------------------------------

__global__ void k_transpose(const float* __restrict__ in, float* __restrict__ out) {
    int i = blockIdx.x * blockDim.x + threadIdx.x;   // over B*N
    if (i >= BATCH * NPTS) return;
    int b = i / NPTS, p = i - b * NPTS;
    for (int d = 0; d < 3; ++d)
        out[(size_t)i * 3 + d] = in[(size_t)b * 3 * NPTS + d * NPTS + p];
}

__global__ void k_fps(const float* __restrict__ xyz, int n, int npoint,
                      int* __restrict__ outIdx) {
    __shared__ float dist[4096];
    __shared__ float rv[256];
    __shared__ int   ri[256];
    __shared__ int   curS;
    int b = blockIdx.x, tid = threadIdx.x;
    const float* base = xyz + (size_t)b * n * 3;
    for (int i = tid; i < n; i += 256) dist[i] = 1e10f;
    if (tid == 0) curS = 0;
    __syncthreads();
    for (int it = 0; it < npoint; ++it) {
        int cur = curS;
        if (tid == 0) outIdx[b * npoint + it] = cur;
        float cx = base[cur * 3], cy = base[cur * 3 + 1], cz = base[cur * 3 + 2];
        float bestV = -1.f; int bestI = 0;
        for (int i = tid; i < n; i += 256) {
            float dx = base[i * 3] - cx, dy = base[i * 3 + 1] - cy, dz = base[i * 3 + 2] - cz;
            float nd = fminf(dist[i], dx * dx + dy * dy + dz * dz);
            dist[i] = nd;
            if (nd > bestV) { bestV = nd; bestI = i; }
        }
        rv[tid] = bestV; ri[tid] = bestI;
        __syncthreads();
        for (int sft = 128; sft > 0; sft >>= 1) {
            if (tid < sft) {
                if (rv[tid + sft] > rv[tid] ||
                    (rv[tid + sft] == rv[tid] && ri[tid + sft] < ri[tid])) {
                    rv[tid] = rv[tid + sft]; ri[tid] = ri[tid + sft];
                }
            }
            __syncthreads();
        }
        if (tid == 0) curS = ri[0];
        __syncthreads();
    }
}

__global__ void k_gather(const float* __restrict__ src, const int* __restrict__ idx,
                         float* __restrict__ out, int npoint, int n, int total) {
    int i = blockIdx.x * blockDim.x + threadIdx.x;   // over B*npoint
    if (i >= total) return;
    int b = i / npoint;
    int j = idx[i];
    for (int d = 0; d < 3; ++d)
        out[(size_t)i * 3 + d] = src[((size_t)b * n + j) * 3 + d];
}

// fold BN: y = (x@W^T + b)*inv*g + beta  ->  acc*s + t ; convert W to padded f16
__global__ void k_convert(const float* __restrict__ W, const float* __restrict__ bias,
                          const float* __restrict__ beta, const float* __restrict__ g,
                          int Cout, int Cin, _Float16* __restrict__ Wp,
                          int Cp, int Kld, float* __restrict__ sOut, float* __restrict__ tOut) {
    int i = blockIdx.x * blockDim.x + threadIdx.x;
    if (i < Cp) {
        float inv = rsqrtf(1.f + 1e-5f);
        float sv = 0.f, tv = 0.f;
        if (i < Cout) { sv = inv * g[i]; tv = bias[i] * inv * g[i] + beta[i]; }
        sOut[i] = sv; tOut[i] = tv;
    }
    if (i >= Cp * Kld) return;
    int co = i / Kld, ci = i - co * Kld;
    float v = (co < Cout && ci < Cin) ? W[(size_t)co * Cin + ci] : 0.f;
    Wp[i] = (_Float16)v;
}

// pack concat [src0(C0) | src1(C1, row = r/rep) | zero pad] as f16 rows
__global__ void k_pack2(const float* __restrict__ s0, int C0,
                        const float* __restrict__ s1, int C1, int rep,
                        _Float16* __restrict__ A, int Kld, int M) {
    int i = blockIdx.x * blockDim.x + threadIdx.x;
    if (i >= M * Kld) return;
    int r = i / Kld, c = i - r * Kld;
    float v = 0.f;
    if (c < C0) v = s0[(size_t)r * C0 + c];
    else if (c < C0 + C1) v = s1[(size_t)(r / rep) * C1 + (c - C0)];
    A[i] = (_Float16)v;
}

// fp1 input: [cls(1) | xyz(3) | xyz(3) | interp(128) | pad] -> 160 cols
__global__ void k_pack_fp1(const float* __restrict__ cls, const float* __restrict__ xyzT,
                           const float* __restrict__ interp, _Float16* __restrict__ A,
                           int Kld, int M) {
    int i = blockIdx.x * blockDim.x + threadIdx.x;
    if (i >= M * Kld) return;
    int r = i / Kld, c = i - r * Kld;
    float v = 0.f;
    if (c == 0) v = cls[r / NPTS];
    else if (c < 4) v = xyzT[(size_t)r * 3 + (c - 1)];
    else if (c < 7) v = xyzT[(size_t)r * 3 + (c - 4)];
    else if (c < 135) v = interp[(size_t)r * 128 + (c - 7)];
    A[i] = (_Float16)v;
}

// 3-NN inverse-distance interpolation
__global__ void k_interp3(const float* __restrict__ xyz1, const float* __restrict__ xyz2,
                          const float* __restrict__ f2, float* __restrict__ out,
                          int n1, int s, int C, int total) {
    int i = blockIdx.x * blockDim.x + threadIdx.x;   // over B*n1
    if (i >= total) return;
    int b = i / n1;
    float px = xyz1[(size_t)i * 3], py = xyz1[(size_t)i * 3 + 1], pz = xyz1[(size_t)i * 3 + 2];
    const float* x2 = xyz2 + (size_t)b * s * 3;
    float d0 = 3.4e38f, d1 = 3.4e38f, d2 = 3.4e38f;
    int i0 = 0, i1 = 0, i2 = 0;
    for (int j = 0; j < s; ++j) {
        float dx = x2[j * 3] - px, dy = x2[j * 3 + 1] - py, dz = x2[j * 3 + 2] - pz;
        float d = dx * dx + dy * dy + dz * dz;
        if (d < d0)      { d2 = d1; i2 = i1; d1 = d0; i1 = i0; d0 = d; i0 = j; }
        else if (d < d1) { d2 = d1; i2 = i1; d1 = d;  i1 = j; }
        else if (d < d2) { d2 = d;  i2 = j; }
    }
    float w0 = 1.f / (d0 + 1e-8f), w1 = 1.f / (d1 + 1e-8f), w2 = 1.f / (d2 + 1e-8f);
    float ws = w0 + w1 + w2; w0 /= ws; w1 /= ws; w2 /= ws;
    const float* fb = f2 + (size_t)b * s * C;
    float* ob = out + (size_t)i * C;
    for (int c = 0; c < C; ++c)
        ob[c] = w0 * fb[(size_t)i0 * C + c] + w1 * fb[(size_t)i1 * C + c] + w2 * fb[(size_t)i2 * C + c];
}

__global__ void k_maxpool(const float* __restrict__ in, int C, int k, int groups,
                          float* __restrict__ out, int outStride, int chOff) {
    int i = blockIdx.x * blockDim.x + threadIdx.x;
    if (i >= groups * C) return;
    int g = i / C, c = i - g * C;
    const float* p = in + (size_t)g * k * C + c;
    float m = p[0];
    for (int r = 1; r < k; ++r) m = fmaxf(m, p[(size_t)r * C]);
    out[(size_t)g * outStride + chOff + c] = m;
}

// head2 (128 -> 5, no bn) + log_softmax
__global__ void k_head2(const _Float16* __restrict__ feat, const float* __restrict__ W2,
                        const float* __restrict__ b2, float* __restrict__ out, int M) {
    int i = blockIdx.x * blockDim.x + threadIdx.x;
    if (i >= M) return;
    const _Float16* f = feat + (size_t)i * 128;
    float acc[5];
#pragma unroll
    for (int j = 0; j < 5; ++j) acc[j] = b2[j];
    for (int c = 0; c < 128; ++c) {
        float fv = (float)f[c];
#pragma unroll
        for (int j = 0; j < 5; ++j) acc[j] += fv * W2[j * 128 + c];
    }
    float m = acc[0];
#pragma unroll
    for (int j = 1; j < 5; ++j) m = fmaxf(m, acc[j]);
    float sum = 0.f;
#pragma unroll
    for (int j = 0; j < 5; ++j) sum += expf(acc[j] - m);
    float l = logf(sum);
#pragma unroll
    for (int j = 0; j < 5; ++j) out[(size_t)i * 5 + j] = acc[j] - m - l;
}

__global__ void k_copy(const float* __restrict__ src, float* __restrict__ dst, int cnt) {
    int i = blockIdx.x * blockDim.x + threadIdx.x;
    if (i < cnt) dst[i] = src[i];
}

// ---------------- Host orchestration ----------------------------------------

static inline int ru(int x, int m) { return (x + m - 1) / m * m; }

extern "C" void kernel_launch(void* const* d_in, const int* in_sizes, int n_in,
                              void* d_out, int out_size, void* d_ws, size_t ws_size,
                              hipStream_t stream) {
    (void)in_sizes; (void)n_in; (void)out_size; (void)ws_size;
    const float* xyz = (const float*)d_in[0];
    const float* cls = (const float*)d_in[1];
    auto P = [&](int i) { return (const float*)d_in[i]; };

    // bump allocator over d_ws
    size_t off = 0;
    auto alloc = [&](size_t bytes) -> void* {
        size_t a = (off + 255) & ~(size_t)255;
        off = a + bytes;
        return (void*)((char*)d_ws + a);
    };

    // ---- scratch buffers
    float* xyzT    = (float*)alloc((size_t)BATCH * NPTS * 3 * 4);
    int*   idx1    = (int*)  alloc((size_t)BATCH * 1024 * 4);
    float* nx1     = (float*)alloc((size_t)BATCH * 1024 * 3 * 4);
    float* l1p     = (float*)alloc((size_t)BATCH * 1024 * 320 * 4);
    int*   idx2    = (int*)  alloc((size_t)BATCH * 128 * 4);
    float* nx2     = (float*)alloc((size_t)BATCH * 128 * 3 * 4);
    float* l2p     = (float*)alloc((size_t)BATCH * 128 * 512 * 4);
    float* l3      = (float*)alloc((size_t)BATCH * 1024 * 4);
    float* mlp3out = (float*)alloc((size_t)BATCH * 128 * 1024 * 4);
    float* l2pNew  = (float*)alloc((size_t)BATCH * 128 * 256 * 4);
    float* interp2 = (float*)alloc((size_t)BATCH * 1024 * 256 * 4);
    float* l1pNew  = (float*)alloc((size_t)BATCH * 1024 * 128 * 4);
    float* interp1 = (float*)alloc((size_t)BATCH * NPTS * 128 * 4);
    _Float16* Abig = (_Float16*)alloc((size_t)32768 * 160 * 2 + (size_t)8192 * 576 * 2);
    _Float16* actP = (_Float16*)alloc((size_t)32768 * 128 * 2 * 2);
    _Float16* actQ = (_Float16*)alloc((size_t)32768 * 128 * 2 * 2);

    // ---- fold & convert weights (params flattened in jax pytree/sorted order)
    auto mk = [&](int base, int Cin, int Cout) -> LayerDesc {
        int Kld = ru(Cin, 32), Cp = ru(Cout, 32);
        _Float16* Wp = (_Float16*)alloc((size_t)Cp * Kld * 2);
        float* sArr = (float*)alloc((size_t)Cp * 4);
        float* tArr = (float*)alloc((size_t)Cp * 4);
        int tot = Cp * Kld;
        k_convert<<<(tot + 255) / 256, 256, 0, stream>>>(
            P(base), P(base + 1), P(base + 2), P(base + 3),
            Cout, Cin, Wp, Cp, Kld, sArr, tArr);
        LayerDesc L; L.W = Wp; L.s = sArr; L.t = tArr;
        L.Kld = Kld; L.CoutPad = Cp; L.CoutReal = Cout;
        return L;
    };

    LayerDesc fp1L[2]  = { mk(2, 135, 128),  mk(6, 128, 128) };
    LayerDesc fp2L[2]  = { mk(10, 576, 256), mk(14, 256, 128) };
    LayerDesc fp3L[2]  = { mk(18, 1536, 256), mk(22, 256, 256) };
    LayerDesc head1L   = mk(26, 128, 128);
    // head2: raw f32 weights d_in[30], bias d_in[31]
    LayerDesc sa1m0[3] = { mk(32, 6, 32),  mk(36, 32, 32), mk(40, 32, 64) };
    LayerDesc sa1m1[3] = { mk(44, 6, 64),  mk(48, 64, 64), mk(52, 64, 128) };
    LayerDesc sa1m2[3] = { mk(56, 6, 64),  mk(60, 64, 96), mk(64, 96, 128) };
    LayerDesc sa2m0[3] = { mk(68, 323, 128), mk(72, 128, 128), mk(76, 128, 256) };
    LayerDesc sa2m1[3] = { mk(80, 323, 128), mk(84, 128, 196), mk(88, 196, 256) };
    LayerDesc sa3L[3]  = { mk(92, 515, 256), mk(96, 256, 512), mk(100, 512, 1024) };

    auto launch_group = [&](const float* xyzP, const float* nxP, const float* ftP,
                            int n, int np, int k, int pc, float radius,
                            LayerDesc* L, float* outF, int os, int co) {
        int sA = L[0].Kld > L[2].Kld ? L[0].Kld : L[2].Kld;
        int sB = L[1].Kld > L[2].CoutPad ? L[1].Kld : L[2].CoutPad;
        size_t smem = (size_t)k * (sA + sB) * 2 + (size_t)k * 4 + 16;
        k_group_mlp<<<BATCH * np, 128, smem, stream>>>(
            xyzP, nxP, ftP, n, np, k, pc, radius * radius,
            L[0], L[1], L[2], outF, os, co, sA, sB);
    };
    auto gemm = [&](const _Float16* A, int lda, int M, LayerDesc L,
                    _Float16* o16, int ldo, float* o32, int st32, int co) {
        int pairs = (M / 16) * (L.CoutPad / 32);
        k_gemm<<<(pairs + 3) / 4, 128, 0, stream>>>(A, lda, M, L, o16, ldo, o32, st32, co);
    };

    // ---- l0: transpose xyz
    k_transpose<<<(BATCH * NPTS + 255) / 256, 256, 0, stream>>>(xyz, xyzT);

    // ---- sa1 (MSG, 3 scales -> l1_points 320ch)
    k_fps<<<BATCH, 256, 0, stream>>>(xyzT, NPTS, 1024, idx1);
    k_gather<<<(BATCH * 1024 + 255) / 256, 256, 0, stream>>>(xyzT, idx1, nx1, 1024, NPTS, BATCH * 1024);
    launch_group(xyzT, nx1, xyzT, NPTS, 1024, 32,  3, 0.1f, sa1m0, l1p, 320, 0);
    launch_group(xyzT, nx1, xyzT, NPTS, 1024, 64,  3, 0.2f, sa1m1, l1p, 320, 64);
    launch_group(xyzT, nx1, xyzT, NPTS, 1024, 128, 3, 0.4f, sa1m2, l1p, 320, 192);

    // ---- sa2 (2 scales -> l2_points 512ch)
    k_fps<<<BATCH, 256, 0, stream>>>(nx1, 1024, 128, idx2);
    k_gather<<<(BATCH * 128 + 255) / 256, 256, 0, stream>>>(nx1, idx2, nx2, 128, 1024, BATCH * 128);
    launch_group(nx1, nx2, l1p, 1024, 128, 64,  320, 0.4f, sa2m0, l2p, 512, 0);
    launch_group(nx1, nx2, l1p, 1024, 128, 128, 320, 0.8f, sa2m1, l2p, 512, 256);

    // ---- sa3: concat(l2_xyz, l2_points) -> MLP [256,512,1024] -> max over 128
    {
        int M = BATCH * 128;
        k_pack2<<<(M * 544 + 255) / 256, 256, 0, stream>>>(nx2, 3, l2p, 512, 1, Abig, 544, M);
        gemm(Abig, 544, M, sa3L[0], actP, 256, nullptr, 0, 0);
        gemm(actP, 256, M, sa3L[1], actQ, 512, nullptr, 0, 0);
        gemm(actQ, 512, M, sa3L[2], nullptr, 0, mlp3out, 1024, 0);
        k_maxpool<<<(BATCH * 1024 + 255) / 256, 256, 0, stream>>>(mlp3out, 1024, 128, BATCH, l3, 1024, 0);
    }

    // ---- fp3: broadcast l3, concat with l2_points -> [256,256]
    {
        int M = BATCH * 128;
        k_pack2<<<(M * 1536 + 255) / 256, 256, 0, stream>>>(l2p, 512, l3, 1024, 128, Abig, 1536, M);
        gemm(Abig, 1536, M, fp3L[0], actP, 256, nullptr, 0, 0);
        gemm(actP, 256, M, fp3L[1], nullptr, 0, l2pNew, 256, 0);
    }

    // ---- fp2: 3-NN interp of l2pNew onto l1 points, concat l1_points -> [256,128]
    {
        int M = BATCH * 1024;
        k_interp3<<<(M + 255) / 256, 256, 0, stream>>>(nx1, nx2, l2pNew, interp2, 1024, 128, 256, M);
        k_pack2<<<(M * 576 + 255) / 256, 256, 0, stream>>>(l1p, 320, interp2, 256, 1, Abig, 576, M);
        gemm(Abig, 576, M, fp2L[0], actP, 256, nullptr, 0, 0);
        gemm(actP, 256, M, fp2L[1], nullptr, 0, l1pNew, 128, 0);
    }

    // ---- fp1: 3-NN interp onto all points, concat [cls|xyz|xyz] -> [128,128]
    {
        int M = BATCH * NPTS;
        k_interp3<<<(M + 255) / 256, 256, 0, stream>>>(xyzT, nx1, l1pNew, interp1, NPTS, 1024, 128, M);
        k_pack_fp1<<<(M * 160 + 255) / 256, 256, 0, stream>>>(cls, xyzT, interp1, Abig, 160, M);
        gemm(Abig, 160, M, fp1L[0], actP, 128, nullptr, 0, 0);
        gemm(actP, 128, M, fp1L[1], actQ, 128, nullptr, 0, 0);
        // head1 (bn+relu)
        gemm(actQ, 128, M, head1L, actP, 128, nullptr, 0, 0);
        // head2 + log_softmax -> output 0
        k_head2<<<(M + 255) / 256, 256, 0, stream>>>(actP, P(30), P(31), (float*)d_out, M);
    }

    // ---- output 1: l3_points (B,1024,1) transposed == flat copy
    k_copy<<<(BATCH * 1024 + 255) / 256, 256, 0, stream>>>(
        l3, (float*)d_out + (size_t)BATCH * NPTS * 5, BATCH * 1024);
}
